// GraphMambaLayer_53257594470660
// MI455X (gfx1250) — matrix operand order repaired and checked by reading, not compiled, not run on hardware
//
#include <hip/hip_runtime.h>

// ---------------------------------------------------------------------------
// GraphMambaLayer collapsed form (Hilbert permutation provably cancels):
//   out = slots + LayerNorm(SiLU(slots @ W1 + b1) @ W2 + b2) * gamma + beta
// bf16 WMMA (v_wmma_f32_16x16x32_bf16), f32 accumulate, fused epilogue.
// Weight chunks staged to LDS by the Tensor Data Mover (tensor_load_to_lds,
// TENSORcnt-tracked, double-buffered, HW-padded rows), compute overlapped.
// Each wave owns 32x128 (GEMM1) / 32x64 (GEMM2): every B fragment feeds 2
// WMMAs; B frags ping-pong with one-iteration lookahead (no WAR hazard NOPs).
// ---------------------------------------------------------------------------

typedef __attribute__((ext_vector_type(16))) __bf16 v16bf;
typedef __attribute__((ext_vector_type(8)))  float  v8f;

#define D_MODEL 256
#define D_INNER 512
#define M_TILE  64

// Padded LDS strides (elements); all fragment chunks stay 16B-aligned.
#define LDA 264   // bf16: X row (256+8)  -> 528 B
#define LDW 40    // bf16: W chunk row (32+8) -> 80 B  (20 dw; 16 lanes hit 16 distinct banks)
#define LDH 520   // bf16: H row (512+8)  -> 1040 B
#define LDY 264   // f32 : Y row (256+8)

#define SZ_A   (M_TILE * LDA * 2)            // 33792
#define OFF_W  (SZ_A)
#define SZ_W   (2 * 512 * LDW * 2)           // 81920 (double-buffered)
#define OFF_H  (OFF_W + SZ_W)                // 115712
#define SZ_H   (M_TILE * LDH * 2)            // 66560
#define OFF_Y  (OFF_H + SZ_H)                // 182272
#define SZ_Y   (M_TILE * LDY * 4)            // 67584
#define SMEM_BYTES (OFF_Y + SZ_Y)            // 249856 B ~ 244 KB (< 320 KB/WGP)

union Frag {
    v16bf v;
    uint4 q[2];
};

#define WMMA_BF16(A, B, C) __builtin_amdgcn_wmma_f32_16x16x32_bf16( \
    false, (A).v, false, (B).v, (short)0, (C), false, false)

#if defined(__has_builtin)
#if __has_builtin(__builtin_amdgcn_tensor_load_to_lds) && \
    __has_builtin(__builtin_amdgcn_s_wait_tensorcnt)
#define USE_TDM 1
#endif
#endif
#ifndef USE_TDM
#define USE_TDM 0
#endif

#if USE_TDM
typedef unsigned int u32x4 __attribute__((ext_vector_type(4)));
typedef int          i32x8 __attribute__((ext_vector_type(8)));
typedef int          i32x4 __attribute__((ext_vector_type(4)));

// 2D bf16 tile load Global->LDS via TDM:
//   rows x 32 elements, global row stride = stride_elems,
//   LDS rows packed at 64 B + 16 B HW padding (pad_interval=16DW, pad_amount=4DW)
//   => LDS row stride 80 B == LDW elements.
__device__ __forceinline__ void tdm_load_w_chunk(const __bf16* gptr,
                                                 unsigned lds_off,
                                                 int rows, int stride_elems)
{
    const unsigned long long ga = (unsigned long long)(const void*)gptr;
    u32x4 g0;
    g0[0] = 1u;                                        // count=1 (valid user D#)
    g0[1] = lds_off;                                   // lds_addr (bytes)
    g0[2] = (unsigned)(ga & 0xffffffffu);              // global_addr[31:0]
    g0[3] = (unsigned)((ga >> 32) & 0x01ffffffu)       // global_addr[56:32]
          | (2u << 30);                                // type = 2 ("image")
    i32x8 g1;
    g1[0] = (1 << 16)                                  // data_size = 2 bytes
          | (1 << 20)                                  // pad_enable
          | (3 << 22)                                  // pad_interval: 16 DWORDs
          | (3 << 25);                                 // pad_amount : 4 DWORDs (16 B)
    g1[1] = (stride_elems & 0xffff) << 16;             // tensor_dim0[15:0] (=row len)
    g1[2] = ((stride_elems >> 16) & 0xffff)            // tensor_dim0[31:16]
          | ((rows & 0xffff) << 16);                   // tensor_dim1[15:0]
    g1[3] = (32 << 16);                                // tile_dim0 = 32
    g1[4] = rows & 0xffff;                             // tile_dim1 = rows (tile_dim2=0)
    g1[5] = stride_elems;                              // tensor_dim0_stride[31:0]
    g1[6] = 0;                                         // stride hi / dim1_stride (unused, 2D)
    g1[7] = 0;
    const i32x4 z4 = {0, 0, 0, 0};                     // groups 2/3 unused (2D tensor)
    const i32x8 z8 = {0, 0, 0, 0, 0, 0, 0, 0};
    __builtin_amdgcn_tensor_load_to_lds(g0, g1, z4, z4, z8, 0);
}
#endif

// ---- one-time weight transpose + f32 -> bf16 conversion (512 KB total) -----
__global__ void prep_weights(const float* __restrict__ W1,
                             const float* __restrict__ W2,
                             __bf16* __restrict__ W1T,   // [D_INNER][D_MODEL]
                             __bf16* __restrict__ W2T)   // [D_MODEL][D_INNER]
{
    const int n   = D_MODEL * D_INNER;     // 131072
    const int idx = blockIdx.x * blockDim.x + threadIdx.x;
    if (idx < n) {
        const int i = idx / D_MODEL;
        const int d = idx % D_MODEL;
        W1T[idx] = (__bf16)W1[d * D_INNER + i];
    } else if (idx < 2 * n) {
        const int j = idx - n;
        const int d = j / D_INNER;
        const int i = j % D_INNER;
        W2T[j] = (__bf16)W2[i * D_MODEL + d];
    }
}

// ---- fused MLP + LayerNorm + residual --------------------------------------
__global__ __launch_bounds__(256)
void fused_mlp_ln(const float*  __restrict__ slots,
                  const __bf16* __restrict__ W1T,
                  const __bf16* __restrict__ W2T,
                  const float*  __restrict__ b1,
                  const float*  __restrict__ b2,
                  const float*  __restrict__ gamma,
                  const float*  __restrict__ beta,
                  float*        __restrict__ out)
{
    extern __shared__ char smem[];
    __bf16* sA = (__bf16*)(smem);
    __bf16* sW = (__bf16*)(smem + OFF_W);
    __bf16* sH = (__bf16*)(smem + OFF_H);
    float*  sY = (float*) (smem + OFF_Y);
    __bf16* const wbuf[2] = { sW, sW + 512 * LDW };

    const int  tid  = threadIdx.x;
    const int  wave = tid >> 5;
    const int  lane = tid & 31;
    const int  hl   = lane >> 4;
    const int  ln16 = lane & 15;
    const long tok0 = (long)blockIdx.x * M_TILE;

    // ---------------- X tile: 64 x 256, f32 -> bf16 -------------------------
    for (int i = tid; i < M_TILE * (D_MODEL / 4); i += 256) {
        const int r  = i / (D_MODEL / 4);
        const int c4 = i % (D_MODEL / 4);
        const float4 v = ((const float4*)(slots + (tok0 + r) * D_MODEL))[c4];
        __bf16* dst = sA + r * LDA + c4 * 4;
        dst[0] = (__bf16)v.x; dst[1] = (__bf16)v.y;
        dst[2] = (__bf16)v.z; dst[3] = (__bf16)v.w;
    }

    // ---------------- GEMM1: H = SiLU(X @ W1 + b1)  [64 x 512] --------------
    // wave grid 2(M) x 4(N); per wave 32x128 = 2x8 accum tiles
    {
        const int mb = (wave >> 2) * 32;
        const int nb = (wave & 3) * 128;
        v8f acc0[8] = {}, acc1[8] = {};

        const int aoff0 = (mb + ln16) * LDA + hl * 8;
        const int aoff1 = (mb + 16 + ln16) * LDA + hl * 8;
        int boff[8];
        #pragma unroll
        for (int t = 0; t < 8; ++t)
            boff[t] = (nb + t * 16 + ln16) * LDW + hl * 8;

        auto g1_step = [&](const __bf16* buf, int kc) {
            Frag a0, a1;
            const __bf16* ap0 = sA + aoff0 + kc;
            const __bf16* ap1 = sA + aoff1 + kc;
            a0.q[0] = *(const uint4*)(ap0); a0.q[1] = *(const uint4*)(ap0 + 16);
            a1.q[0] = *(const uint4*)(ap1); a1.q[1] = *(const uint4*)(ap1 + 16);
            Frag bp[2];
            bp[0].q[0] = *(const uint4*)(buf + boff[0]);
            bp[0].q[1] = *(const uint4*)(buf + boff[0] + 16);
            #pragma unroll
            for (int t = 0; t < 8; ++t) {
                if (t + 1 < 8) {                 // lookahead load (ping-pong)
                    const __bf16* bn = buf + boff[t + 1];
                    bp[(t + 1) & 1].q[0] = *(const uint4*)(bn);
                    bp[(t + 1) & 1].q[1] = *(const uint4*)(bn + 16);
                }
                acc0[t] = WMMA_BF16(a0, bp[t & 1], acc0[t]);
                acc1[t] = WMMA_BF16(a1, bp[t & 1], acc1[t]);
            }
        };

#if USE_TDM
        if (wave == 0)
            tdm_load_w_chunk(W1T, (unsigned)(size_t)(void*)wbuf[0], 512, D_MODEL);
        for (int c = 0; c < 8; ++c) {
            if (wave == 0) __builtin_amdgcn_s_wait_tensorcnt(0);
            __syncthreads();                          // chunk c visible to all waves
            if (wave == 0 && c + 1 < 8)               // prefetch next chunk (overlaps WMMA)
                tdm_load_w_chunk(W1T + (c + 1) * 32,
                                 (unsigned)(size_t)(void*)wbuf[(c + 1) & 1],
                                 512, D_MODEL);
            g1_step(wbuf[c & 1], c * 32);
        }
        if (wave == 0)                                // overlap with SiLU epilogue
            tdm_load_w_chunk(W2T, (unsigned)(size_t)(void*)wbuf[0], 256, D_INNER);
#else
        for (int c = 0; c < 8; ++c) {
            for (int i = tid; i < 512 * 4; i += 256) {
                const int n = i >> 2, seg = i & 3;
                *(uint4*)(sW + n * LDW + seg * 8) =
                    *(const uint4*)(W1T + n * D_MODEL + c * 32 + seg * 8);
            }
            __syncthreads();
            g1_step(sW, c * 32);
            __syncthreads();
        }
#endif

        // bias + SiLU -> sH (C/D layout: VGPR g -> row 8*hl+g, lane%16 -> col)
        #pragma unroll
        for (int t = 0; t < 8; ++t) {
            const int   col  = nb + t * 16 + ln16;
            const float bias = b1[col];
            #pragma unroll
            for (int g = 0; g < 8; ++g) {
                const int   r0 = mb + hl * 8 + g;
                const float v0 = acc0[t][g] + bias;
                sH[r0 * LDH + col] = (__bf16)(v0 / (1.0f + __expf(-v0)));
                const int   r1 = mb + 16 + hl * 8 + g;
                const float v1 = acc1[t][g] + bias;
                sH[r1 * LDH + col] = (__bf16)(v1 / (1.0f + __expf(-v1)));
            }
        }
    }
#if !USE_TDM
    __syncthreads();
#endif

    // ---------------- GEMM2: Y = H @ W2 + b2  [64 x 256] --------------------
    // wave grid 2(M) x 4(N); per wave 32x64 = 2x4 accum tiles
    {
        const int mb = (wave >> 2) * 32;
        const int nb = (wave & 3) * 64;
        v8f acc0[4] = {}, acc1[4] = {};

        const int aoff0 = (mb + ln16) * LDH + hl * 8;
        const int aoff1 = (mb + 16 + ln16) * LDH + hl * 8;
        int boff[4];
        #pragma unroll
        for (int t = 0; t < 4; ++t)
            boff[t] = (nb + t * 16 + ln16) * LDW + hl * 8;

        auto g2_step = [&](const __bf16* buf, int kc) {
            Frag a0, a1;
            const __bf16* ap0 = sH + aoff0 + kc;
            const __bf16* ap1 = sH + aoff1 + kc;
            a0.q[0] = *(const uint4*)(ap0); a0.q[1] = *(const uint4*)(ap0 + 16);
            a1.q[0] = *(const uint4*)(ap1); a1.q[1] = *(const uint4*)(ap1 + 16);
            Frag bp[2];
            bp[0].q[0] = *(const uint4*)(buf + boff[0]);
            bp[0].q[1] = *(const uint4*)(buf + boff[0] + 16);
            #pragma unroll
            for (int t = 0; t < 4; ++t) {
                if (t + 1 < 4) {
                    const __bf16* bn = buf + boff[t + 1];
                    bp[(t + 1) & 1].q[0] = *(const uint4*)(bn);
                    bp[(t + 1) & 1].q[1] = *(const uint4*)(bn + 16);
                }
                acc0[t] = WMMA_BF16(a0, bp[t & 1], acc0[t]);
                acc1[t] = WMMA_BF16(a1, bp[t & 1], acc1[t]);
            }
        };

#if USE_TDM
        for (int c = 0; c < 16; ++c) {
            if (wave == 0) __builtin_amdgcn_s_wait_tensorcnt(0);
            __syncthreads();                          // c==0 barrier also publishes sH
            if (wave == 0 && c + 1 < 16)
                tdm_load_w_chunk(W2T + (c + 1) * 32,
                                 (unsigned)(size_t)(void*)wbuf[(c + 1) & 1],
                                 256, D_INNER);
            g2_step(wbuf[c & 1], c * 32);
        }
#else
        for (int c = 0; c < 16; ++c) {
            for (int i = tid; i < 256 * 4; i += 256) {
                const int n = i >> 2, seg = i & 3;
                *(uint4*)(sW + n * LDW + seg * 8) =
                    *(const uint4*)(W2T + n * D_INNER + c * 32 + seg * 8);
            }
            __syncthreads();
            g2_step(sW, c * 32);
            __syncthreads();
        }
#endif

        #pragma unroll
        for (int t = 0; t < 4; ++t) {
            const int   col  = nb + t * 16 + ln16;
            const float bias = b2[col];
            #pragma unroll
            for (int g = 0; g < 8; ++g) {
                const int r0 = mb + hl * 8 + g;
                sY[r0 * LDY + col] = acc0[t][g] + bias;
                const int r1 = mb + 16 + hl * 8 + g;
                sY[r1 * LDY + col] = acc1[t][g] + bias;
            }
        }
    }
    __syncthreads();

    // ---------------- LayerNorm + residual, f32 out -------------------------
    // each wave normalizes 8 rows; 32 lanes x 8 elems per row
    for (int rr = 0; rr < 8; ++rr) {
        const int    row = wave * 8 + rr;
        const float* yr  = sY + row * LDY;
        float vals[8];
        float sum = 0.0f, sq = 0.0f;
        #pragma unroll
        for (int j = 0; j < 8; ++j) {
            const float v = yr[lane + 32 * j];
            vals[j] = v; sum += v; sq += v * v;
        }
        #pragma unroll
        for (int s = 16; s > 0; s >>= 1) {
            sum += __shfl_xor(sum, s, 32);
            sq  += __shfl_xor(sq,  s, 32);
        }
        const float mu   = sum * (1.0f / 256.0f);
        const float var  = sq * (1.0f / 256.0f) - mu * mu;
        const float rstd = rsqrtf(var + 1e-5f);
        const long  base = (tok0 + row) * D_MODEL;
        #pragma unroll
        for (int j = 0; j < 8; ++j) {
            const int col = lane + 32 * j;
            out[base + col] = slots[base + col]
                            + (vals[j] - mu) * rstd * gamma[col] + beta[col];
        }
    }
}

// ---------------------------------------------------------------------------
extern "C" void kernel_launch(void* const* d_in, const int* in_sizes, int n_in,
                              void* d_out, int out_size, void* d_ws, size_t ws_size,
                              hipStream_t stream) {
    const float* slots = (const float*)d_in[0];
    // d_in[1] = adj (unused), d_in[2] = centroids (permutation cancels)
    const float* W1    = (const float*)d_in[3];
    const float* b1    = (const float*)d_in[4];
    const float* W2    = (const float*)d_in[5];
    const float* b2    = (const float*)d_in[6];
    const float* gamma = (const float*)d_in[7];
    const float* beta  = (const float*)d_in[8];

    __bf16* W1T = (__bf16*)d_ws;
    __bf16* W2T = W1T + D_MODEL * D_INNER;

    const int nW = 2 * D_MODEL * D_INNER;
    prep_weights<<<(nW + 255) / 256, 256, 0, stream>>>(W1, W2, W1T, W2T);

    const int tokens = in_sizes[0] / D_MODEL;       // B*K = 131072
    fused_mlp_ln<<<tokens / M_TILE, 256, SMEM_BYTES, stream>>>(
        slots, W1T, W2T, b1, b2, gamma, beta, (float*)d_out);
}
